// CoReMultiheadAttention_18829136625790
// MI455X (gfx1250) — matrix operand order repaired and checked by reading, not compiled, not run on hardware
//
#include <hip/hip_runtime.h>
#include <hip/hip_bf16.h>

// ---- problem constants (from reference) ----
#define BB   16
#define EE   1024
#define HH   16
#define NN   2048
#define EH   64        // E/H
#define BH_  256       // B*H
#define NB_  32768     // N*B  (rows of the out-proj GEMM)
#define NM1  2047      // N-1

typedef __attribute__((ext_vector_type(16))) __bf16 v16bf;
typedef __attribute__((ext_vector_type(8)))  __bf16 v8bf;
typedef __attribute__((ext_vector_type(8)))  float  v8f;

// ---------------------------------------------------------------------------
// f32 -> bf16 conversion
// ---------------------------------------------------------------------------
__global__ void f32_to_bf16_kernel(const float* __restrict__ src,
                                   __bf16* __restrict__ dst, int n) {
    int i = blockIdx.x * blockDim.x + threadIdx.x;
    if (i < n) dst[i] = (__bf16)src[i];
}

// ---------------------------------------------------------------------------
// extract k_old[bh][e] = k_t_mem[bh, e, 0]   (strided gather, done once)
// ---------------------------------------------------------------------------
__global__ void extract_k_old_kernel(const float* __restrict__ kt,
                                     float* __restrict__ k_old) {
    int i = blockIdx.x * 256 + threadIdx.x;          // i < BH_*EH = 16384
    int bh = i >> 6, e = i & 63;
    k_old[i] = kt[(size_t)bh * EH * NN + (size_t)e * NN];
}

// ---------------------------------------------------------------------------
__device__ __forceinline__ v8f wmma_bf16_acc(v16bf a, v16bf b, v8f c) {
    // (neg_a, A, neg_b, B, c_mod, C, reuse_a, reuse_b)
    return __builtin_amdgcn_wmma_f32_16x16x32_bf16(false, a, false, b,
                                                   (short)0, c, false, false);
}

// ---------------------------------------------------------------------------
// Direct-load wave-level bf16 WMMA GEMM (used for the small in-projections):
// C[m,n] = scale * ( sum_k A[m,k]*Bw[n,k] + bias[n] ),  C = A * Bw^T.
// One wave -> 16x64 slab. Block = 8 waves; total_waves = (M/16)*(N/64).
// ---------------------------------------------------------------------------
__global__ __launch_bounds__(256)
void wmma_gemm_bf16_kernel(const __bf16* __restrict__ A,
                           const __bf16* __restrict__ Bw,
                           const float*  __restrict__ bias,
                           float* __restrict__ C,
                           int M, int N, int K, float scale) {
    const int lane = threadIdx.x & 31;
    const int gw   = blockIdx.x * 8 + (threadIdx.x >> 5);
    const int mtiles = M >> 4;
    const int mt = gw % mtiles;
    const int ng = gw / mtiles;
    const int m0 = mt << 4;
    const int n0 = ng << 6;
    if (n0 >= N) return;                 // wave-uniform

    const int rowA  = m0 + (lane & 15);
    const int kAoff = (lane >> 4) << 4;
    const int colB  = n0 + (lane & 15);
    const int kBoff = (lane >> 4) << 4;

    const __bf16* pA = A  + (size_t)rowA * K + kAoff;
    const __bf16* pB = Bw + (size_t)colB * K + kBoff;

    v8f acc0 = {}, acc1 = {}, acc2 = {}, acc3 = {};

    for (int k0 = 0; k0 < K; k0 += 32) {
        v8bf a_lo = *(const v8bf*)(pA + k0);
        v8bf a_hi = *(const v8bf*)(pA + k0 + 8);
        v16bf afrag;
        #pragma unroll
        for (int i = 0; i < 8; ++i) { afrag[i] = a_lo[i]; afrag[i + 8] = a_hi[i]; }

        v16bf b0 = *(const v16bf*)(pB + k0);
        v16bf b1 = *(const v16bf*)(pB + (size_t)16 * K + k0);
        v16bf b2 = *(const v16bf*)(pB + (size_t)32 * K + k0);
        v16bf b3 = *(const v16bf*)(pB + (size_t)48 * K + k0);

        __builtin_prefetch(pB + k0 + 32, 0, 1);
        __builtin_prefetch(pA + k0 + 32, 0, 1);

        acc0 = wmma_bf16_acc(afrag, b0, acc0);
        acc1 = wmma_bf16_acc(afrag, b1, acc1);
        acc2 = wmma_bf16_acc(afrag, b2, acc2);
        acc3 = wmma_bf16_acc(afrag, b3, acc3);
    }

    const int mrow  = (lane >> 4) << 3;
    const int nbase = n0 + (lane & 15);
    const float bi0 = bias[nbase];
    const float bi1 = bias[nbase + 16];
    const float bi2 = bias[nbase + 32];
    const float bi3 = bias[nbase + 48];
    #pragma unroll
    for (int r = 0; r < 8; ++r) {
        float* crow = C + (size_t)(m0 + mrow + r) * N;
        crow[nbase]      = scale * (acc0[r] + bi0);
        crow[nbase + 16] = scale * (acc1[r] + bi1);
        crow[nbase + 32] = scale * (acc2[r] + bi2);
        crow[nbase + 48] = scale * (acc3[r] + bi3);
    }
}

// ---------------------------------------------------------------------------
// LDS-staged bf16 WMMA GEMM for the big out-projection (K must be 1024).
// All 8 waves of a block share one 64-column B panel; the panel (64 x 1024
// bf16 = 128 KB) is staged into LDS once via gfx1250 async global->LDS DMA
// (tracked by ASYNCcnt), then the K loop reads fragments with ds_load_b128.
// Column stride padded to 2064 B (516 dwords == 4 mod 64) -> the 16 lanes of
// a b128 fragment fetch start on distinct 4-bank groups.
// Block computes 128(M) x 64(N); grid = (M/128)*(N/64).
// ---------------------------------------------------------------------------
#define BPAN_COLSTRIDE 2064
#define BPAN_BYTES     (64 * BPAN_COLSTRIDE)   // 132096 B of dynamic LDS

__global__ __launch_bounds__(256)
void wmma_gemm_lds_bf16_kernel(const __bf16* __restrict__ A,
                               const __bf16* __restrict__ Bw,
                               const float*  __restrict__ bias,
                               float* __restrict__ C,
                               int M, int N, int K) {
    extern __shared__ char bpan[];
    const int lane = threadIdx.x & 31;
    const int wave = threadIdx.x >> 5;
    const int mtb  = blockIdx.x % (M >> 7);
    const int ng   = blockIdx.x / (M >> 7);
    const int m0   = ((mtb << 3) + wave) << 4;
    const int n0   = ng << 6;

    // ---- stage B panel: 64 cols x 2048 B = 8192 x 16B chunks, 1024 per wave,
    //      32 chunks (one per lane) per async instruction, 32 instructions/wave.
    {
        const uint32_t lds_base = (uint32_t)(uintptr_t)bpan;
        const char* gbase = (const char*)(Bw + (size_t)n0 * K);
        for (int it = 0; it < 32; ++it) {
            const int g   = ((wave << 5) + it) * 32 + lane;  // 0..8191
            const int col = g >> 7;                          // 128 chunks per col
            const int kc  = (g & 127) << 4;                  // byte offset in col
            const uint32_t dstoff = lds_base + col * BPAN_COLSTRIDE + kc;
            const void* src = gbase + (size_t)col * (size_t)(K * 2) + kc;
            asm volatile("global_load_async_to_lds_b128 %0, %1, off"
                         :: "v"(dstoff), "v"(src) : "memory");
        }
        asm volatile("s_wait_asynccnt 0x0" ::: "memory");
        __syncthreads();
    }

    const int rowA  = m0 + (lane & 15);
    const int kAoff = (lane >> 4) << 4;
    const __bf16* pA = A + (size_t)rowA * K + kAoff;

    const int cB   = lane & 15;
    const int kseg = (lane >> 4) << 5;                       // 0 or 32 bytes
    const char* lp = bpan + cB * BPAN_COLSTRIDE + kseg;

    v8f acc0 = {}, acc1 = {}, acc2 = {}, acc3 = {};

    for (int k0 = 0; k0 < K; k0 += 32) {
        v8bf a_lo = *(const v8bf*)(pA + k0);
        v8bf a_hi = *(const v8bf*)(pA + k0 + 8);
        v16bf afrag;
        #pragma unroll
        for (int i = 0; i < 8; ++i) { afrag[i] = a_lo[i]; afrag[i + 8] = a_hi[i]; }

        const int kb = k0 << 1;
        v16bf b0 = *(const v16bf*)(lp + kb);
        v16bf b1 = *(const v16bf*)(lp + 16 * BPAN_COLSTRIDE + kb);
        v16bf b2 = *(const v16bf*)(lp + 32 * BPAN_COLSTRIDE + kb);
        v16bf b3 = *(const v16bf*)(lp + 48 * BPAN_COLSTRIDE + kb);

        __builtin_prefetch(pA + k0 + 32, 0, 1);

        acc0 = wmma_bf16_acc(afrag, b0, acc0);
        acc1 = wmma_bf16_acc(afrag, b1, acc1);
        acc2 = wmma_bf16_acc(afrag, b2, acc2);
        acc3 = wmma_bf16_acc(afrag, b3, acc3);
    }

    const int mrow  = (lane >> 4) << 3;
    const int nbase = n0 + (lane & 15);
    const float bi0 = bias[nbase];
    const float bi1 = bias[nbase + 16];
    const float bi2 = bias[nbase + 32];
    const float bi3 = bias[nbase + 48];
    #pragma unroll
    for (int r = 0; r < 8; ++r) {
        float* crow = C + (size_t)(m0 + mrow + r) * N;
        crow[nbase]      = acc0[r] + bi0;
        crow[nbase + 16] = acc1[r] + bi1;
        crow[nbase + 32] = acc2[r] + bi2;
        crow[nbase + 48] = acc3[r] + bi3;
    }
}

// ---------------------------------------------------------------------------
// sdpa top rows: one wave per (bh, j), j = 0..N-2.
// ---------------------------------------------------------------------------
__global__ __launch_bounds__(256)
void sdpa_top_kernel(const float* __restrict__ q_mem,
                     const float* __restrict__ k_old,
                     const float* __restrict__ k_vec,
                     const float* __restrict__ v_vec,
                     const float* __restrict__ v_mem,
                     const float* __restrict__ av_mem,
                     const float* __restrict__ a_sum_mem,
                     __bf16* __restrict__ attn,
                     float* __restrict__ a_sum_out,
                     float* __restrict__ av_out,
                     float* __restrict__ q_out) {
    const int lane = threadIdx.x & 31;
    const int gw   = blockIdx.x * 8 + (threadIdx.x >> 5);   // < BH_*NM1
    const int bh   = gw / NM1;
    const int j    = gw % NM1;
    const int e0 = lane, e1 = lane + 32;

    const size_t qrow = ((size_t)bh * NN + (j + 1)) * EH;
    const float q0 = q_mem[qrow + e0];
    const float q1 = q_mem[qrow + e1];

    const int hb = bh * EH;
    float d0 = q0 * k_old[hb + e0] + q1 * k_old[hb + e1];
    float d1 = q0 * k_vec[hb + e0] + q1 * k_vec[hb + e1];
    #pragma unroll
    for (int m = 16; m >= 1; m >>= 1) {
        d0 += __shfl_xor(d0, m, 32);
        d1 += __shfl_xor(d1, m, 32);
    }
    const float a0 = __expf(d0);
    const float a1 = __expf(d1);
    const float s  = a_sum_mem[(size_t)bh * NM1 + j] + a1 - a0;

    const size_t avrow = ((size_t)bh * NM1 + j) * EH;
    const float vo0 = v_mem[(size_t)bh * (NN * EH) + e0];   // v_mem[bh,0,e]
    const float vo1 = v_mem[(size_t)bh * (NN * EH) + e1];
    const float av0 = av_mem[avrow + e0] - a0 * vo0 + a1 * v_vec[hb + e0];
    const float av1 = av_mem[avrow + e1] - a0 * vo1 + a1 * v_vec[hb + e1];

    const float inv = 1.0f / s;
    const int b = bh >> 4, h = bh & 15;
    const size_t arow = ((size_t)j * BB + b) * EE + h * EH;
    attn[arow + e0] = (__bf16)(av0 * inv);
    attn[arow + e1] = (__bf16)(av1 * inv);

    const size_t qorow = ((size_t)bh * NN + j) * EH;        // shifted q state
    q_out[qorow + e0] = q0;
    q_out[qorow + e1] = q1;

    if (j >= 1) {
        const size_t srow = (size_t)bh * NM1 + (j - 1);
        if (lane == 0) a_sum_out[srow] = s;
        av_out[srow * EH + e0] = av0;
        av_out[srow * EH + e1] = av1;
    }
}

// ---------------------------------------------------------------------------
// sdpa bottom row: one block (256 threads) per bh.
// ---------------------------------------------------------------------------
__global__ __launch_bounds__(256)
void sdpa_bottom_kernel(const float* __restrict__ q_vec,
                        const float* __restrict__ k_vec,
                        const float* __restrict__ v_vec,
                        const float* __restrict__ k_t,
                        const float* __restrict__ v_mem,
                        __bf16* __restrict__ attn,
                        float* __restrict__ a_sum_out,
                        float* __restrict__ av_out,
                        float* __restrict__ q_out,
                        float* __restrict__ v_out) {
    __shared__ float qs[EH], vs[EH], av_acc[EH], sum_acc, dbr_acc;
    const int bh  = blockIdx.x;
    const int tid = threadIdx.x;

    if (tid < EH) {
        qs[tid] = q_vec[bh * EH + tid];
        vs[tid] = v_vec[bh * EH + tid];
        av_acc[tid] = 0.f;
    }
    if (tid == 0) { sum_acc = 0.f; dbr_acc = 0.f; }
    __syncthreads();
    if (tid < EH) atomicAdd(&dbr_acc, qs[tid] * k_vec[bh * EH + tid]);

    float av_part[EH];
    #pragma unroll
    for (int e = 0; e < EH; ++e) av_part[e] = 0.f;
    float lsum = 0.f;

    const float* ktb = k_t + (size_t)bh * EH * NN;
    for (int s = tid; s < NM1; s += 256) {
        float d = 0.f;
        #pragma unroll
        for (int e = 0; e < EH; ++e) d += qs[e] * ktb[(size_t)e * NN + s + 1];
        const float a = __expf(d);
        lsum += a;
        const float* vrow = v_mem + ((size_t)bh * NN + s + 1) * EH;
        #pragma unroll
        for (int e = 0; e < EH; ++e) av_part[e] += a * vrow[e];
    }
    atomicAdd(&sum_acc, lsum);                    // ds_add_f32
    #pragma unroll
    for (int e = 0; e < EH; ++e) atomicAdd(&av_acc[e], av_part[e]);
    __syncthreads();

    if (tid < EH) {
        const float a_br = __expf(dbr_acc);
        const float asum = sum_acc + a_br;
        const float avb  = av_acc[tid] + a_br * vs[tid];
        const int b = bh >> 4, h = bh & 15;
        attn[((size_t)NM1 * BB + b) * EE + h * EH + tid] = (__bf16)(avb / asum);
        if (tid == 0) a_sum_out[(size_t)bh * NM1 + (NM1 - 1)] = asum;
        av_out[((size_t)bh * NM1 + (NM1 - 1)) * EH + tid] = avb;
        q_out[((size_t)bh * NN + (NN - 1)) * EH + tid] = qs[tid];   // q_step
        v_out[((size_t)bh * NN + (NN - 1)) * EH + tid] = vs[tid];   // v_step
    }
}

// ---------------------------------------------------------------------------
// shifted copy for v_mem state (float4; last row written by sdpa_bottom)
// ---------------------------------------------------------------------------
__global__ void shift_rows_v_kernel(const float* __restrict__ src,
                                    float* __restrict__ dst) {
    const size_t i4 = (size_t)blockIdx.x * blockDim.x + threadIdx.x; // < BH_*NN*EH/4
    const int r = (int)((i4 >> 4) & (NN - 1));
    if (r == NN - 1) return;
    ((float4*)dst)[i4] = ((const float4*)src)[i4 + 16];
}

// ---------------------------------------------------------------------------
// shifted copy for k_t state: along last dim (last col = k_step)
// ---------------------------------------------------------------------------
__global__ void shift_k_kernel(const float* __restrict__ kt,
                               const float* __restrict__ k_vec,
                               float* __restrict__ dst) {
    const size_t i = (size_t)blockIdx.x * blockDim.x + threadIdx.x; // < BH_*EH*NN
    const int pos = (int)(i & (NN - 1));
    dst[i] = (pos < NN - 1) ? kt[i + 1] : k_vec[i >> 11];           // i>>11 = bh*EH+e
}

// ---------------------------------------------------------------------------
extern "C" void kernel_launch(void* const* d_in, const int* in_sizes, int n_in,
                              void* d_out, int out_size, void* d_ws, size_t ws_size,
                              hipStream_t stream) {
    const float* query      = (const float*)d_in[0];
    const float* key_in     = (const float*)d_in[1];
    const float* value      = (const float*)d_in[2];
    const float* a_sum_mem  = (const float*)d_in[3];
    const float* av_mem     = (const float*)d_in[4];
    const float* q_mem      = (const float*)d_in[5];
    const float* k_t_mem    = (const float*)d_in[6];
    const float* v_mem      = (const float*)d_in[7];
    const float* w_in       = (const float*)d_in[8];
    const float* b_in       = (const float*)d_in[9];
    const float* w_out      = (const float*)d_in[10];
    const float* b_out      = (const float*)d_in[11];

    // ---- workspace carve (~75.8 MB) ----
    char* w = (char*)d_ws;
    __bf16* attn_bf  = (__bf16*)w; w += (size_t)NB_ * EE * 2;        // 64 MiB
    __bf16* w_in_bf  = (__bf16*)w; w += (size_t)3 * EE * EE * 2;     // 6 MiB
    __bf16* w_out_bf = (__bf16*)w; w += (size_t)EE * EE * 2;         // 2 MiB
    __bf16* qkv_bf   = (__bf16*)w; w += (size_t)3 * BB * EE * 2;
    float*  q_vec    = (float*)w;  w += (size_t)BB * EE * 4;
    float*  k_vec    = (float*)w;  w += (size_t)BB * EE * 4;
    float*  v_vec    = (float*)w;  w += (size_t)BB * EE * 4;
    float*  k_old    = (float*)w;  w += (size_t)BH_ * EH * 4;
    (void)ws_size; (void)in_sizes; (void)n_in; (void)out_size;

    // ---- output carve (concatenated tuple) ----
    float* out       = (float*)d_out;                                 // (N,B,E)
    float* a_sum_out = out + (size_t)NN * BB * EE;                    // (BH, N-1)
    float* av_out    = a_sum_out + (size_t)BH_ * NM1;                 // (BH, N-1, Eh)
    float* q_out     = av_out + (size_t)BH_ * NM1 * EH;               // (BH, N, Eh)
    float* k_out     = q_out + (size_t)BH_ * NN * EH;                 // (BH, Eh, N)
    float* v_out     = k_out + (size_t)BH_ * EH * NN;                 // (BH, N, Eh)

    // 1. bf16 conversions for GEMM operands
    f32_to_bf16_kernel<<<(BB * EE + 255) / 256, 256, 0, stream>>>(query,  qkv_bf,               BB * EE);
    f32_to_bf16_kernel<<<(BB * EE + 255) / 256, 256, 0, stream>>>(key_in, qkv_bf + BB * EE,     BB * EE);
    f32_to_bf16_kernel<<<(BB * EE + 255) / 256, 256, 0, stream>>>(value,  qkv_bf + 2 * BB * EE, BB * EE);
    f32_to_bf16_kernel<<<(3 * EE * EE) / 256, 256, 0, stream>>>(w_in,  w_in_bf,  3 * EE * EE);
    f32_to_bf16_kernel<<<(EE * EE) / 256, 256, 0, stream>>>(w_out, w_out_bf, EE * EE);

    // 2. k_old gather
    extract_k_old_kernel<<<BH_ * EH / 256, 256, 0, stream>>>(k_t_mem, k_old);

    // 3. in-projections via WMMA (M=16, N=1024, K=1024); q folds 1/sqrt(Eh).
    wmma_gemm_bf16_kernel<<<2, 256, 0, stream>>>(qkv_bf,               w_in_bf,               b_in,           q_vec, BB, EE, EE, 0.125f);
    wmma_gemm_bf16_kernel<<<2, 256, 0, stream>>>(qkv_bf + BB * EE,     w_in_bf + EE * EE,     b_in + EE,      k_vec, BB, EE, EE, 1.0f);
    wmma_gemm_bf16_kernel<<<2, 256, 0, stream>>>(qkv_bf + 2 * BB * EE, w_in_bf + 2 * EE * EE, b_in + 2 * EE,  v_vec, BB, EE, EE, 1.0f);

    // 4. attention state update
    sdpa_top_kernel<<<BH_ * NM1 / 8, 256, 0, stream>>>(q_mem, k_old, k_vec, v_vec, v_mem,
                                                       av_mem, a_sum_mem,
                                                       attn_bf, a_sum_out, av_out, q_out);
    sdpa_bottom_kernel<<<BH_, 256, 0, stream>>>(q_vec, k_vec, v_vec, k_t_mem, v_mem,
                                                attn_bf, a_sum_out, av_out, q_out, v_out);

    // 5. shifted state copies
    shift_rows_v_kernel<<<(BH_ * NN * EH / 4) / 256, 256, 0, stream>>>(v_mem, v_out);
    shift_k_kernel<<<(BH_ * EH * NN) / 256, 256, 0, stream>>>(k_t_mem, k_vec, k_out);

    // 6. out-projection via async-LDS-staged WMMA:
    //    grid = (M/128)*(N/64) = 256*16 = 4096 blocks, 128 KB B-panel in LDS.
    wmma_gemm_lds_bf16_kernel<<<4096, 256, BPAN_BYTES, stream>>>(
        attn_bf, w_out_bf, b_out, out, NB_, EE, EE);
}
